// DynLocRep_loss_45921790329145
// MI455X (gfx1250) — matrix operand order repaired and checked by reading, not compiled, not run on hardware
//
#include <hip/hip_runtime.h>
#include <hip/hip_bf16.h>

typedef __attribute__((ext_vector_type(16))) _Float16 v16h;
typedef __attribute__((ext_vector_type(8)))  _Float16 v8h;
typedef __attribute__((ext_vector_type(8)))  float    v8f;

#define BSZ   2048
#define NVIEW 2
#define NTOT  4096     // BSZ * NVIEW
#define DDIM  256
#define INVT  (1.0f / 0.07f)

// ---------------------------------------------------------------------------
// Kernel 1: reindex [B,V,D] fp32 -> view-major [N,D] f16, and expand labels.
// f[v*B + b][d] = features[b][v][d];  lab2[i] = labels[i % B]
// ---------------------------------------------------------------------------
__global__ __launch_bounds__(256) void prep_kernel(
    const float* __restrict__ features, const int* __restrict__ labels,
    _Float16* __restrict__ fh, int* __restrict__ lab2) {
  const int gid = blockIdx.x * 256 + threadIdx.x;          // 0 .. N*D-1
  const int i = gid >> 8;          // row in [0, N)
  const int d = gid & (DDIM - 1);
  const int v = i >> 11;           // i / BSZ
  const int b = i & (BSZ - 1);     // i % BSZ
  fh[gid] = (_Float16)features[(b * NVIEW + v) * DDIM + d];
  if (gid < NTOT) lab2[gid] = labels[gid & (BSZ - 1)];
}

// ---------------------------------------------------------------------------
// 16x16 Gram tile via 8 chained v_wmma_f32_16x16x32_f16 over K = 256.
// A fragments are preloaded; B fragments streamed per tile.
// B layout (32x16, K-major per lane): lanes 0-15 hold K=q*32+{0-7,16-23} of column
// n = colBase + lane; lanes 16-31 hold K=q*32+{8-15,24-31}.
// ---------------------------------------------------------------------------
__device__ __forceinline__ v8f gram_tile(const v16h a[8],
                                         const _Float16* __restrict__ fh,
                                         int colBase, int lane16, int hi) {
  v8f c = {};
  const _Float16* colp = fh + (size_t)(colBase + lane16) * DDIM + hi * 8;
#pragma unroll
  for (int q = 0; q < 8; ++q) {
    union { v16h v; v8h h[2]; } bu;
    bu.h[0] = *(const v8h*)(colp + q * 32);
    bu.h[1] = *(const v8h*)(colp + q * 32 + 16);
    c = __builtin_amdgcn_wmma_f32_16x16x32_f16(
        /*neg_a=*/false, a[q], /*neg_b=*/false, bu.v,
        /*c_mod=*/(short)0, c, /*reuse_a=*/false, /*reuse_b=*/false);
  }
  return c;
}

__device__ __forceinline__ float redmax16(float v) {
#pragma unroll
  for (int m = 1; m < 16; m <<= 1) v = fmaxf(v, __shfl_xor(v, m, 32));
  return v;
}
__device__ __forceinline__ float redsum16(float v) {
#pragma unroll
  for (int m = 1; m < 16; m <<= 1) v += __shfl_xor(v, m, 32);
  return v;
}

// ---------------------------------------------------------------------------
// Kernel 2: one wave per 16-row strip. Two column sweeps:
//   sweep 1 -> per-row max of logits; sweep 2 -> S (sum exp, no diag),
//   T (sum posmask * shifted-logit), P (positive count).
// ---------------------------------------------------------------------------
__global__ __launch_bounds__(32) void supcon_main(
    const _Float16* __restrict__ fh, const int* __restrict__ lab2,
    float* __restrict__ Srow, float* __restrict__ Trow,
    float* __restrict__ Prow) {
  const int lane   = threadIdx.x;       // 0..31, one wave32
  const int lane16 = lane & 15;
  const int hi     = lane >> 4;
  const int rowBase = blockIdx.x * 16;  // 256 strips

  // Preload A fragments: 16 rows x 256 K, ISA 16-bit A layout.
  v16h a[8];
  const _Float16* rowp = fh + (size_t)(rowBase + lane16) * DDIM;
#pragma unroll
  for (int q = 0; q < 8; ++q) {
    union { v16h v; v8h h[2]; } au;
    au.h[0] = *(const v8h*)(rowp + q * 32 + hi * 8);        // K q*32 + {0-7 | 8-15}
    au.h[1] = *(const v8h*)(rowp + q * 32 + 16 + hi * 8);   // K q*32 + {16-23 | 24-31}
    a[q] = au.v;
  }

  // Row labels for this lane's 8 C-layout rows (row = rowBase + r + 8*hi).
  int labrow[8];
#pragma unroll
  for (int r = 0; r < 8; ++r) labrow[r] = lab2[rowBase + r + 8 * hi];

  // ---- Sweep 1: row max (in raw dot units; scale by 1/tau afterwards) ----
  float rm[8];
#pragma unroll
  for (int r = 0; r < 8; ++r) rm[r] = -3.4e38f;
  for (int ct = 0; ct < NTOT / 16; ++ct) {
    v8f c = gram_tile(a, fh, ct * 16, lane16, hi);
#pragma unroll
    for (int r = 0; r < 8; ++r) rm[r] = fmaxf(rm[r], c[r]);
  }
#pragma unroll
  for (int r = 0; r < 8; ++r) rm[r] = redmax16(rm[r]) * INVT;

  // ---- Sweep 2: masked exp-sums with the shift applied ----
  float S[8], T[8], P[8];
#pragma unroll
  for (int r = 0; r < 8; ++r) { S[r] = 0.0f; T[r] = 0.0f; P[r] = 0.0f; }

  for (int ct = 0; ct < NTOT / 16; ++ct) {
    const int colBase = ct * 16;
    v8f c = gram_tile(a, fh, colBase, lane16, hi);
    const int n    = colBase + lane16;   // this lane's column
    const int labn = lab2[n];
#pragma unroll
    for (int r = 0; r < 8; ++r) {
      const int row  = rowBase + r + 8 * hi;
      const float l  = c[r] * INVT - rm[r];
      const bool  dg = (n == row);
      S[r] += dg ? 0.0f : __expf(l);
      const bool pm = (!dg) && (labn == labrow[r]);
      T[r] += pm ? l : 0.0f;
      P[r] += pm ? 1.0f : 0.0f;
    }
  }

#pragma unroll
  for (int r = 0; r < 8; ++r) {
    S[r] = redsum16(S[r]);
    T[r] = redsum16(T[r]);
    P[r] = redsum16(P[r]);
  }
  if (lane16 == 0) {
#pragma unroll
    for (int r = 0; r < 8; ++r) {
      const int row = rowBase + r + 8 * hi;
      Srow[row] = S[r];
      Trow[row] = T[r];
      Prow[row] = P[r];
    }
  }
}

// ---------------------------------------------------------------------------
// Kernel 3: final scalar reduction. loss = -(sum_i T_i - P_i*log(S_i)) / sum_i P_i
// ---------------------------------------------------------------------------
__global__ __launch_bounds__(256) void finalize_kernel(
    const float* __restrict__ Srow, const float* __restrict__ Trow,
    const float* __restrict__ Prow, float* __restrict__ out) {
  __shared__ float snum[256];
  __shared__ float sden[256];
  const int tid = threadIdx.x;
  float num = 0.0f, den = 0.0f;
  for (int i = tid; i < NTOT; i += 256) {
    num += Trow[i] - Prow[i] * __logf(Srow[i]);
    den += Prow[i];
  }
  snum[tid] = num;
  sden[tid] = den;
  __syncthreads();
  for (int s = 128; s > 0; s >>= 1) {
    if (tid < s) { snum[tid] += snum[tid + s]; sden[tid] += sden[tid + s]; }
    __syncthreads();
  }
  if (tid == 0) out[0] = -snum[0] / sden[0];   // (TEMP/BASE_TEMP) == 1
}

// ---------------------------------------------------------------------------
extern "C" void kernel_launch(void* const* d_in, const int* in_sizes, int n_in,
                              void* d_out, int out_size, void* d_ws, size_t ws_size,
                              hipStream_t stream) {
  const float* features = (const float*)d_in[0];   // [2048, 2, 256] fp32
  const int*   labels   = (const int*)d_in[1];     // [2048] int32
  float*       out      = (float*)d_out;           // scalar fp32

  // Workspace layout (all 256B-aligned):
  //   fh   : N*D f16      = 2 MB
  //   lab2 : N int32      = 16 KB
  //   Srow / Trow / Prow  : 3 * 16 KB
  char* ws = (char*)d_ws;
  _Float16* fh   = (_Float16*)ws;
  int*      lab2 = (int*)(ws + (size_t)NTOT * DDIM * sizeof(_Float16));
  float*    Srow = (float*)((char*)lab2 + NTOT * sizeof(int));
  float*    Trow = Srow + NTOT;
  float*    Prow = Trow + NTOT;

  // 1) fp32 -> f16 reindexed copy + label expansion
  prep_kernel<<<(NTOT * DDIM) / 256, 256, 0, stream>>>(features, labels, fh, lab2);

  // 2) WMMA Gram sweeps: one wave per 16-row strip
  supcon_main<<<NTOT / 16, 32, 0, stream>>>(fh, lab2, Srow, Trow, Prow);

  // 3) scalar loss
  finalize_kernel<<<1, 256, 0, stream>>>(Srow, Trow, Prow, out);
}
